// VectorQuantizerEMA_16217796510394
// MI455X (gfx1250) — matrix operand order repaired and checked by reading, not compiled, not run on hardware
//
#include <hip/hip_runtime.h>
#include <hip/hip_bf16.h>
#include <stdint.h>

// Problem constants (from reference): z_e (32,1024,64), embedding (4096,64)
#define N_PTS   32768
#define DIM     64
#define KCB     4096
#define EPS_F   1e-5f

#define CHUNK_COLS 32                 // codewords staged per LDS buffer (32*64*2B = 4KB)
#define NCHUNK     (KCB / CHUNK_COLS) // 128

typedef __attribute__((ext_vector_type(8)))  _Float16 v8h;
typedef __attribute__((ext_vector_type(16))) _Float16 v16h;
typedef __attribute__((ext_vector_type(8)))  float    v8f;
typedef __attribute__((ext_vector_type(4)))  int      v4i;

#if __has_builtin(__builtin_amdgcn_global_load_async_to_lds_b128)
#define HAVE_ASYNC_LDS 1
typedef __attribute__((address_space(1))) v4i g_v4i;   // global int4
typedef __attribute__((address_space(3))) v4i l_v4i;   // LDS int4
#endif

__device__ __forceinline__ void stage16(const _Float16* g, _Float16* l) {
#ifdef HAVE_ASYNC_LDS
  // async DMA: global -> LDS, 16B per lane, tracked with ASYNCcnt
  __builtin_amdgcn_global_load_async_to_lds_b128(
      (g_v4i*)(uintptr_t)g, (l_v4i*)l, 0, 0);
#else
  *(v8h*)l = *(const v8h*)g;
#endif
}

__device__ __forceinline__ void stage_wait() {
#ifdef HAVE_ASYNC_LDS
#if __has_builtin(__builtin_amdgcn_s_wait_asynccnt)
  __builtin_amdgcn_s_wait_asynccnt(0);
#else
  asm volatile("s_wait_asynccnt 0x0" ::: "memory");
#endif
#endif
}

__device__ __forceinline__ v16h ld16(const _Float16* p0, const _Float16* p1) {
  v8h lo = *(const v8h*)p0;
  v8h hi = *(const v8h*)p1;
  return __builtin_shufflevector(lo, hi, 0,1,2,3,4,5,6,7,8,9,10,11,12,13,14,15);
}

__device__ __forceinline__ float wave_sum32(float v) {
  #pragma unroll
  for (int off = 1; off < 32; off <<= 1) v += __shfl_xor(v, off, 32);
  return v;
}

// ---------------------------------------------------------------- zero ws accumulators
__global__ void k_zero(float* usage, float* embed_sum) {
  int i = blockIdx.x * blockDim.x + threadIdx.x;
  if (i < KCB * DIM) embed_sum[i] = 0.0f;
  if (i < KCB)       usage[i]     = 0.0f;
}

// ---------------------------------------------------------------- normalize rows of z_e
__global__ __launch_bounds__(256) void k_normalize(const float* __restrict__ z_e,
                                                   float* __restrict__ fn,
                                                   _Float16* __restrict__ fnh) {
  int lane = threadIdx.x & 31;
  int row  = blockIdx.x * 8 + (threadIdx.x >> 5);
  const float2 v = ((const float2*)(z_e + row * DIM))[lane];
  float ss = wave_sum32(v.x * v.x + v.y * v.y);
  float s  = 1.0f / fmaxf(sqrtf(ss), 1e-8f);
  float2 o; o.x = v.x * s; o.y = v.y * s;
  ((float2*)(fn + row * DIM))[lane] = o;
  fnh[row * DIM + 2 * lane]     = (_Float16)o.x;
  fnh[row * DIM + 2 * lane + 1] = (_Float16)o.y;
}

// ---------------------------------------------------------------- embedding f32 -> f16
__global__ void k_conv_emb(const float* __restrict__ emb, _Float16* __restrict__ embh) {
  int i = blockIdx.x * blockDim.x + threadIdx.x;
  if (i < KCB * DIM) embh[i] = (_Float16)emb[i];
}

// ---------------------------------------------------------------- fused GEMM + argmax + epilogue
// Each wave owns a 16-row tile of flat_norm. All 8 waves in the block scan the
// same codebook columns, so B is staged once per block into LDS via
// double-buffered async-to-LDS DMA (ASYNCcnt) and consumed with ds_load_b128.
__global__ __launch_bounds__(256) void k_gemm_argmax(
    const _Float16* __restrict__ fnh, const _Float16* __restrict__ embh,
    const float* __restrict__ fn, const float* __restrict__ emb_f32,
    float* __restrict__ usage, float* __restrict__ embed_sum,
    float* __restrict__ zq_st, float* __restrict__ zq, float* __restrict__ idx_out) {
  __shared__ _Float16 smem[2][CHUNK_COLS * DIM];   // 2 x 4KB

  const int tid  = threadIdx.x;
  const int lane = tid & 31;
  const int m    = lane & 15;       // row-in-tile / col-in-tile slot
  const int hi   = lane >> 4;       // lane half
  const int row0 = (blockIdx.x * 8 + (tid >> 5)) * 16;

  // A fragments (16x64 f16), loaded once, reused for all 256 column tiles.
  const _Float16* arow = fnh + (size_t)(row0 + m) * DIM;
  const v16h a0 = ld16(arow + hi * 8,      arow + hi * 8 + 16);      // features 0..31
  const v16h a1 = ld16(arow + 32 + hi * 8, arow + 32 + hi * 8 + 16); // features 32..63

  float best[8];
  int   bidx[8];
  #pragma unroll
  for (int r = 0; r < 8; ++r) { best[r] = -3.4e38f; bidx[r] = 0; }

  // prologue: stage chunk 0 (each thread DMAs 16B; 256*16B = 4KB = one chunk)
  stage16(embh + (size_t)tid * 8, &smem[0][tid * 8]);
  stage_wait();
  __syncthreads();

  for (int c = 0; c < NCHUNK; ++c) {
    if (c + 1 < NCHUNK)   // prefetch next chunk into the other buffer
      stage16(embh + (size_t)(c + 1) * CHUNK_COLS * DIM + tid * 8,
              &smem[(c + 1) & 1][tid * 8]);

    const _Float16* sb = smem[c & 1];
    #pragma unroll
    for (int sub = 0; sub < 2; ++sub) {
      const int col0 = c * CHUNK_COLS + sub * 16;
      const _Float16* brow = sb + (sub * 16 + m) * DIM + hi * 8;
      const v16h b0 = ld16(brow,      brow + 16);  // features 0..31 (table order, matches a0)
      const v16h b1 = ld16(brow + 32, brow + 48);  // features 32..63 (table order, matches a1)

      v8f cc = {};
      cc = __builtin_amdgcn_wmma_f32_16x16x32_f16(false, a0, false, b0, (short)0, cc, false, false);
      cc = __builtin_amdgcn_wmma_f32_16x16x32_f16(false, a1, false, b1, (short)0, cc, false, false);

      #pragma unroll
      for (int r = 0; r < 8; ++r) {     // cc[r]: row (r + 8*hi), col (col0 + m)
        float v = cc[r];
        if (v > best[r]) { best[r] = v; bidx[r] = col0 + m; }
      }
    }

    stage_wait();       // next buffer's DMA done
    __syncthreads();    // all waves done reading current buffer
  }

  // argmax across the 16 column slots (lanes within each half); tie -> lower index
  #pragma unroll
  for (int off = 1; off < 16; off <<= 1) {
    #pragma unroll
    for (int r = 0; r < 8; ++r) {
      float ov = __shfl_xor(best[r], off, 32);
      int   oi = __shfl_xor(bidx[r], off, 32);
      if (ov > best[r] || (ov == best[r] && oi < bidx[r])) { best[r] = ov; bidx[r] = oi; }
    }
  }

  if (m == 0) {
    #pragma unroll
    for (int r = 0; r < 8; ++r) idx_out[row0 + r + 8 * hi] = (float)bidx[r];
  }

  // epilogue: gather z_q rows, scatter usage / embed_sum (embedding rows are L2-hot)
  #pragma unroll
  for (int mm = 0; mm < 16; ++mm) {
    const int k   = __shfl(bidx[mm & 7], (mm < 8) ? 0 : 16, 32);
    const int n_g = row0 + mm;
    float2 e = ((const float2*)(emb_f32 + (size_t)k * DIM))[lane];
    ((float2*)(zq    + (size_t)n_g * DIM))[lane] = e;
    ((float2*)(zq_st + (size_t)n_g * DIM))[lane] = e;
    float2 f = ((const float2*)(fn + (size_t)n_g * DIM))[lane];
    atomicAdd(&embed_sum[(size_t)k * DIM + 2 * lane],     f.x);
    atomicAdd(&embed_sum[(size_t)k * DIM + 2 * lane + 1], f.y);
    if (lane == 0) atomicAdd(&usage[k], 1.0f);
  }
}

// ---------------------------------------------------------------- EMA buffers
__global__ void k_ema(const float* __restrict__ ema_emb, const float* __restrict__ embed_sum,
                      const float* __restrict__ ema_cs, const float* __restrict__ usage,
                      float* __restrict__ new_ema, float* __restrict__ new_cs) {
  const float d = 0.99f, od = 1.0f - 0.99f;
  int i = blockIdx.x * blockDim.x + threadIdx.x;
  if (i < KCB * DIM) new_ema[i] = d * ema_emb[i] + od * embed_sum[i];
  if (i < KCB)       new_cs[i]  = d * ema_cs[i]  + od * usage[i];
}

// ---------------------------------------------------------------- threefry2x32 (JAX key(1))
__device__ __forceinline__ uint32_t rotl32(uint32_t x, int r) { return (x << r) | (x >> (32 - r)); }

__device__ void threefry2x32(uint32_t x0, uint32_t x1, uint32_t* o0, uint32_t* o1) {
  const uint32_t ks0 = 0u, ks1 = 1u, ks2 = 0x1BD11BDAu ^ 0u ^ 1u;
  const int R0[4] = {13, 15, 26, 6}, R1[4] = {17, 29, 16, 24};
  x0 += ks0; x1 += ks1;
  #pragma unroll
  for (int i = 0; i < 4; ++i) { x0 += x1; x1 = rotl32(x1, R0[i]); x1 ^= x0; } x0 += ks1; x1 += ks2 + 1u;
  #pragma unroll
  for (int i = 0; i < 4; ++i) { x0 += x1; x1 = rotl32(x1, R1[i]); x1 ^= x0; } x0 += ks2; x1 += ks0 + 2u;
  #pragma unroll
  for (int i = 0; i < 4; ++i) { x0 += x1; x1 = rotl32(x1, R0[i]); x1 ^= x0; } x0 += ks0; x1 += ks1 + 3u;
  #pragma unroll
  for (int i = 0; i < 4; ++i) { x0 += x1; x1 = rotl32(x1, R1[i]); x1 ^= x0; } x0 += ks1; x1 += ks2 + 4u;
  #pragma unroll
  for (int i = 0; i < 4; ++i) { x0 += x1; x1 = rotl32(x1, R0[i]); x1 ^= x0; } x0 += ks2; x1 += ks0 + 5u;
  *o0 = x0; *o1 = x1;
}

// ---------------------------------------------------------------- stats + n + rand_idx (1 block)
__global__ __launch_bounds__(1024) void k_reduce(const float* __restrict__ new_cs,
                                                 const float* __restrict__ usage,
                                                 float* __restrict__ stats,
                                                 float* __restrict__ nsum,
                                                 int* __restrict__ randidx) {
  __shared__ float sA[1024], sB[1024], sC[1024];
  int tid = threadIdx.x;
  float cs_sum = 0.0f, ent = 0.0f, dead = 0.0f;
  for (int k = tid; k < KCB; k += 1024) {
    cs_sum += new_cs[k];
    float u = usage[k];
    if (u > 0.0f) { float p = u * (1.0f / (float)N_PTS); ent -= p * logf(p); }
    else dead += 1.0f;
  }
  sA[tid] = cs_sum; sB[tid] = ent; sC[tid] = dead;
  __syncthreads();
  for (int s = 512; s > 0; s >>= 1) {
    if (tid < s) { sA[tid] += sA[tid + s]; sB[tid] += sB[tid + s]; sC[tid] += sC[tid + s]; }
    __syncthreads();
  }
  if (tid == 0) {
    stats[0] = expf(sB[0]);              // perplexity
    stats[1] = sC[0] / (float)KCB;       // dead ratio
    nsum[0]  = sA[0];                    // sum(new_cs)
  }
  // jax.random.randint(key(1), (K,), 0, N): bits = threefry2x32(iota); span=2^15 -> & 0x7FFF
  for (int p = tid; p < KCB / 2; p += 1024) {
    uint32_t b0, b1;
    threefry2x32((uint32_t)p, (uint32_t)(p + KCB / 2), &b0, &b1);
    randidx[p]           = (int)(b0 & (uint32_t)(N_PTS - 1));
    randidx[p + KCB / 2] = (int)(b1 & (uint32_t)(N_PTS - 1));
  }
}

// ---------------------------------------------------------------- final embedding (wave per k)
__global__ __launch_bounds__(256) void k_finalize(const float* __restrict__ new_cs,
                                                  const float* __restrict__ new_ema,
                                                  const float* __restrict__ usage,
                                                  const float* __restrict__ fn,
                                                  const int* __restrict__ randidx,
                                                  const float* __restrict__ nsum,
                                                  float* __restrict__ new_embedding) {
  int lane = threadIdx.x & 31;
  int k    = blockIdx.x * 8 + (threadIdx.x >> 5);
  float n = nsum[0];
  float smoothed = (new_cs[k] + EPS_F) / (n + (float)KCB * EPS_F);
  float denom = fmaxf(smoothed, EPS_F);
  float2 e = ((const float2*)(new_ema + (size_t)k * DIM))[lane];
  e.x /= denom; e.y /= denom;
  float ss = wave_sum32(e.x * e.x + e.y * e.y);
  float sc = 1.0f / fmaxf(sqrtf(ss), 1e-8f);
  float2 out; out.x = e.x * sc; out.y = e.y * sc;
  if (usage[k] <= 0.0f) {                      // dead-code reinit (uniform per wave)
    int ri = randidx[k];
    float2 s = ((const float2*)(fn + (size_t)ri * DIM))[lane];
    float ss2 = wave_sum32(s.x * s.x + s.y * s.y);
    float sc2 = 1.0f / fmaxf(sqrtf(ss2), 1e-8f);
    out.x = s.x * sc2; out.y = s.y * sc2;
  }
  ((float2*)(new_embedding + (size_t)k * DIM))[lane] = out;
}

// ---------------------------------------------------------------- launch
extern "C" void kernel_launch(void* const* d_in, const int* in_sizes, int n_in,
                              void* d_out, int out_size, void* d_ws, size_t ws_size,
                              hipStream_t stream) {
  const float* z_e     = (const float*)d_in[0];   // (32,1024,64)
  const float* emb     = (const float*)d_in[1];   // (4096,64)
  const float* ema_cs  = (const float*)d_in[2];   // (4096,)
  const float* ema_emb = (const float*)d_in[3];   // (4096,64)

  // outputs concatenated flat in return order (f32)
  float* out      = (float*)d_out;
  float* o_zqst   = out;                                   // 2,097,152
  float* o_zq     = out + 2097152;                         // 2,097,152
  float* o_idx    = out + 4194304;                         // 32,768
  float* o_stats  = out + 4227072;                         // 2
  float* o_newemb = out + 4227074;                         // 262,144
  float* o_newcs  = out + 4489218;                         // 4,096
  float* o_newema = out + 4493314;                         // 262,144

  // workspace carve-up (~13.8 MB)
  float*    fn        = (float*)d_ws;                      // N*D f32
  _Float16* fnh       = (_Float16*)(fn + N_PTS * DIM);     // N*D f16
  _Float16* embh      = fnh + N_PTS * DIM;                 // K*D f16
  float*    usage     = (float*)(embh + KCB * DIM);        // K
  float*    embed_sum = usage + KCB;                       // K*D
  int*      randidx   = (int*)(embed_sum + KCB * DIM);     // K
  float*    nsum      = (float*)(randidx + KCB);           // 1

  k_zero<<<(KCB * DIM + 255) / 256, 256, 0, stream>>>(usage, embed_sum);
  k_normalize<<<N_PTS / 8, 256, 0, stream>>>(z_e, fn, fnh);
  k_conv_emb<<<(KCB * DIM + 255) / 256, 256, 0, stream>>>(emb, embh);
  k_gemm_argmax<<<N_PTS / 128, 256, 0, stream>>>(fnh, embh, fn, emb,
                                                 usage, embed_sum, o_zqst, o_zq, o_idx);
  k_ema<<<(KCB * DIM + 255) / 256, 256, 0, stream>>>(ema_emb, embed_sum, ema_cs, usage,
                                                     o_newema, o_newcs);
  k_reduce<<<1, 1024, 0, stream>>>(o_newcs, usage, o_stats, nsum, randidx);
  k_finalize<<<KCB / 8, 256, 0, stream>>>(o_newcs, o_newema, usage, fn, randidx, nsum, o_newemb);
}
